// Model_67336497266788
// MI455X (gfx1250) — matrix operand order repaired and checked by reading, not compiled, not run on hardware
//
#include <hip/hip_runtime.h>
#include <hip/hip_bf16.h>

typedef __attribute__((ext_vector_type(16))) __bf16 bf16x16;
typedef __attribute__((ext_vector_type(8)))  __bf16 bf16x8;
typedef __attribute__((ext_vector_type(8)))  float  f32x8;

#define NODES 12000
#define NPAD  12032      // 12000 rounded up to 64
#define NEDGE 384000
#define INF   256

// ---------------- elementwise helpers ----------------

__global__ void k_cvt_bf16(const float* __restrict__ in, __bf16* __restrict__ out, int n) {
  int i = blockIdx.x * blockDim.x + threadIdx.x;
  if (i < n) out[i] = (__bf16)in[i];
}

__global__ void k_deg_init(float* __restrict__ deg, int n) {
  int i = blockIdx.x * blockDim.x + threadIdx.x;
  if (i < n) deg[i] = 1.0f;   // self-loop weight
}

__global__ void k_deg_scatter(const int* __restrict__ dst, const float* __restrict__ ew,
                              float* __restrict__ deg, int E) {
  int e = blockIdx.x * blockDim.x + threadIdx.x;
  if (e < E) atomicAdd(&deg[dst[e]], ew[e]);
}

__global__ void k_deg_rsqrt(float* __restrict__ deg, int n) {
  int i = blockIdx.x * blockDim.x + threadIdx.x;
  if (i < n) { float d = deg[i]; deg[i] = (d > 0.0f) ? rsqrtf(d) : 0.0f; }
}

// agg = dinv[i]^2 * xw[i]   (self-loop contribution; also initializes agg)
__global__ void k_agg_self(const float* __restrict__ xw, const float* __restrict__ dinv,
                           float* __restrict__ agg, int n, int C) {
  int idx = blockIdx.x * blockDim.x + threadIdx.x;
  if (idx >= n * C) return;
  int i = idx / C;
  float s = dinv[i];
  agg[idx] = s * s * xw[idx];
}

// scatter: agg[dst] += dinv[s]*w*dinv[d] * xw[src], 4 channels per thread
__global__ void k_agg_edges(const int* __restrict__ src, const int* __restrict__ dst,
                            const float* __restrict__ ew, const float* __restrict__ dinv,
                            const float* __restrict__ xw, float* __restrict__ agg,
                            int E, int C) {
  long long t = (long long)blockIdx.x * blockDim.x + threadIdx.x;
  int c4 = C >> 2;
  long long tot = (long long)E * c4;
  if (t >= tot) return;
  int e = (int)(t / c4);
  int c = (int)(t % c4) * 4;
  int s = src[e], d = dst[e];
  float coef = dinv[s] * ew[e] * dinv[d];
  const float4 xv = *(const float4*)(xw + (size_t)s * C + c);
  float* ap = agg + (size_t)d * C + c;
  atomicAdd(ap + 0, coef * xv.x);
  atomicAdd(ap + 1, coef * xv.y);
  atomicAdd(ap + 2, coef * xv.z);
  atomicAdd(ap + 3, coef * xv.w);
}

// out = bf16( relu(agg + bias[c]) )
__global__ void k_bias_relu_bf16(const float* __restrict__ agg, const float* __restrict__ bias,
                                 __bf16* __restrict__ out, int n, int C) {
  int idx = blockIdx.x * blockDim.x + threadIdx.x;
  if (idx >= n * C) return;
  int c = idx % C;
  float v = agg[idx] + bias[c];
  out[idx] = (__bf16)fmaxf(v, 0.0f);
}

// out[k][j] = in[j][k] with zero padding: in [M][Kc] -> out [Kc][Mpad]
__global__ void k_transpose_pad_bf16(const __bf16* __restrict__ in, __bf16* __restrict__ out,
                                     int M, int Mpad, int Kc) {
  int idx = blockIdx.x * blockDim.x + threadIdx.x;
  if (idx >= Kc * Mpad) return;
  int k = idx / Mpad, j = idx % Mpad;
  out[(size_t)k * Mpad + j] = (j < M) ? in[(size_t)j * Kc + k] : (__bf16)0.0f;
}

// ---------------- WMMA GEMM ----------------
// MLP=false : C[M,N](f32)  =  A[M,K](bf16) @ B[K,Nld](bf16)
// MLP=true  : Cb[M,N](bf16) = relu( A @ B + bias )
// Block: 256 threads = 8 waves (4 x 2). Block tile 128x64, wave tile 32x32, K step 32.
// Double-buffered LDS; A tile via global_load_async_to_lds_b128 (ASYNCcnt),
// B tile staged transposed; one barrier per K step.
template <bool MLP>
__global__ __launch_bounds__(256)
void k_gemm_wmma(const __bf16* __restrict__ A, const __bf16* __restrict__ B,
                 const float* __restrict__ bias, float* __restrict__ C,
                 __bf16* __restrict__ Cb, int M, int N, int Nld, int K) {
  __shared__ __align__(16) __bf16 As[2][128][40];   // [buf][m][k], pad 8
  __shared__ __align__(16) __bf16 Bs[2][64][40];    // [buf][n][k] transposed, pad 8

  const int tid  = threadIdx.x;
  const int wave = tid >> 5;
  const int lane = tid & 31;
  const int mw = (wave & 3) * 32;
  const int nw = (wave >> 2) * 32;
  const int m0 = blockIdx.x * 128;
  const int n0 = blockIdx.y * 64;

  f32x8 acc00 = {}, acc01 = {}, acc10 = {}, acc11 = {};

  // A staging: 2 threads per row, 16 bf16 (two async b128) each
  const int arow  = tid >> 1;
  const int ahalf = (tid & 1) * 16;
  int ag = m0 + arow; if (ag >= M) ag = M - 1;        // clamp; masked at store
  const __bf16* Aptr = A + (size_t)ag * K;
  const unsigned lds_a   = (unsigned)(size_t)&As[0][arow][ahalf];
  const unsigned ABUF_SZ = 128 * 40 * 2;              // bytes per A buffer

  // B staging: thread -> (k row, 8 cols), scatter transposed into LDS
  const int bk = tid >> 3;             // 0..31
  const int bn = (tid & 7) * 8;        // 0..56

  // fragment addressing (per ISA 16-bit A/B layouts)
  const int frow = lane & 15;
  const int fkbA = (lane >> 4) * 8;    // A: lanes 16-31 hold K+8 half
  const int fkbB = (lane >> 4) * 8;    // B: lanes 16-31 hold K+8 half

  auto stage = [&](int k0, int b) {
    const __bf16* g0 = Aptr + k0 + ahalf;
    const unsigned la = lds_a + (b ? ABUF_SZ : 0u);
    asm volatile("global_load_async_to_lds_b128 %0, %1, off"
                 :: "v"(la), "v"((unsigned long long)(size_t)g0) : "memory");
    asm volatile("global_load_async_to_lds_b128 %0, %1, off"
                 :: "v"(la + 16u), "v"((unsigned long long)(size_t)(g0 + 8)) : "memory");
    bf16x8 bv = *(const bf16x8*)(B + (size_t)(k0 + bk) * Nld + (n0 + bn));
    #pragma unroll
    for (int i = 0; i < 8; ++i) Bs[b][bn + i][bk] = bv[i];
  };

  stage(0, 0);
  const int steps = K >> 5;
  for (int i = 0; i < steps; ++i) {
    asm volatile("s_wait_asynccnt 0" ::: "memory");   // A tile i resident in LDS
    __syncthreads();                                  // B tile i visible; old readers done
    if (i + 1 < steps) stage((i + 1) << 5, (i + 1) & 1);

    const int b = i & 1;
    bf16x8 a0lo = *(const bf16x8*)&As[b][mw + frow][fkbA];
    bf16x8 a0hi = *(const bf16x8*)&As[b][mw + frow][16 + fkbA];
    bf16x16 a0 = __builtin_shufflevector(a0lo, a0hi,
        0,1,2,3,4,5,6,7,8,9,10,11,12,13,14,15);
    bf16x8 a1lo = *(const bf16x8*)&As[b][mw + 16 + frow][fkbA];
    bf16x8 a1hi = *(const bf16x8*)&As[b][mw + 16 + frow][16 + fkbA];
    bf16x16 a1 = __builtin_shufflevector(a1lo, a1hi,
        0,1,2,3,4,5,6,7,8,9,10,11,12,13,14,15);

    bf16x8 b0lo = *(const bf16x8*)&Bs[b][nw + frow][fkbB];
    bf16x8 b0hi = *(const bf16x8*)&Bs[b][nw + frow][16 + fkbB];
    bf16x16 b0 = __builtin_shufflevector(b0lo, b0hi,
        0,1,2,3,4,5,6,7,8,9,10,11,12,13,14,15);
    bf16x8 b1lo = *(const bf16x8*)&Bs[b][nw + 16 + frow][fkbB];
    bf16x8 b1hi = *(const bf16x8*)&Bs[b][nw + 16 + frow][16 + fkbB];
    bf16x16 b1 = __builtin_shufflevector(b1lo, b1hi,
        0,1,2,3,4,5,6,7,8,9,10,11,12,13,14,15);

    acc00 = __builtin_amdgcn_wmma_f32_16x16x32_bf16(false, a0, false, b0,
                                                    (short)0, acc00, false, false);
    acc01 = __builtin_amdgcn_wmma_f32_16x16x32_bf16(false, a0, false, b1,
                                                    (short)0, acc01, false, false);
    acc10 = __builtin_amdgcn_wmma_f32_16x16x32_bf16(false, a1, false, b0,
                                                    (short)0, acc10, false, false);
    acc11 = __builtin_amdgcn_wmma_f32_16x16x32_bf16(false, a1, false, b1,
                                                    (short)0, acc11, false, false);
  }

  // epilogue: C/D layout — VGPR r: M = r + (lane>=16 ? 8 : 0), N = lane%16
  const int mh = (lane >> 4) * 8;
  #pragma unroll
  for (int mt = 0; mt < 2; ++mt) {
    #pragma unroll
    for (int nt = 0; nt < 2; ++nt) {
      const f32x8 acc = (mt == 0) ? (nt == 0 ? acc00 : acc01)
                                  : (nt == 0 ? acc10 : acc11);
      int gn = n0 + nw + nt * 16 + frow;
      if (gn >= N) continue;
      float bv = MLP ? bias[gn] : 0.0f;
      #pragma unroll
      for (int r = 0; r < 8; ++r) {
        int gm = m0 + mw + mt * 16 + mh + r;
        if (gm >= M) continue;
        float v = acc[r] + bv;
        if (MLP) {
          Cb[(size_t)gm * N + gn] = (__bf16)fmaxf(v, 0.0f);
        } else {
          C[(size_t)gm * N + gn] = v;
        }
      }
    }
  }
}

// ---------------- host orchestration ----------------

static inline int ceil_div(long long a, int b) { return (int)((a + b - 1) / b); }

extern "C" void kernel_launch(void* const* d_in, const int* in_sizes, int n_in,
                              void* d_out, int out_size, void* d_ws, size_t ws_size,
                              hipStream_t stream) {
  (void)in_sizes; (void)n_in; (void)out_size; (void)ws_size;

  // workspace (small, persistent until final GEMM)
  char* ws = (char*)d_ws;
  __bf16* emb_dis = (__bf16*)(ws);                 // 12000*64 bf16 = 1,536,000 B
  __bf16* drugT   = (__bf16*)(ws + 1600000);       // 64*12032 bf16 = 1,540,096 B
  float*  deg     = (float*) (ws + 3200000);       // 48,000 B
  __bf16* wbuf    = (__bf16*)(ws + 3264000);       // 237,568 bf16 = 475,136 B

  // big scratch lives inside d_out (576 MB); all dead before final GEMM writes
  char* ob = (char*)d_out;
  float*  xw   = (float*) (ob);                    // 12000*256 f32
  float*  agg  = (float*) (ob + 16777216);         // 12000*256 f32
  __bf16* act  = (__bf16*)(ob + 33554432);         // 12000*256 bf16
  __bf16* act2 = (__bf16*)(ob + 41943040);         // 12000*256 bf16

  // weight bf16 sub-offsets (elements)
  __bf16* g1 = wbuf + 0;
  __bf16* g2 = wbuf + 65536;
  __bf16* l1 = wbuf + 131072;
  __bf16* l2 = wbuf + 196608;
  __bf16* l3 = wbuf + 229376;

  const int threads = 256;
  const int nC = NODES * INF;                      // 3,072,000

  auto gemm_f32 = [&](const __bf16* A, const __bf16* B, float* Cf,
                      int M, int N, int Nld, int K) {
    dim3 grid(ceil_div(M, 128), Nld / 64);
    k_gemm_wmma<false><<<grid, dim3(256), 0, stream>>>(A, B, nullptr, Cf, nullptr,
                                                       M, N, Nld, K);
  };
  auto gemm_mlp = [&](const __bf16* A, const __bf16* B, const float* bias, __bf16* Cbf,
                      int M, int N, int Nld, int K) {
    dim3 grid(ceil_div(M, 128), Nld / 64);
    k_gemm_wmma<true><<<grid, dim3(256), 0, stream>>>(A, B, bias, nullptr, Cbf,
                                                      M, N, Nld, K);
  };

  for (int br = 0; br < 2; ++br) {
    const float* x  = (const float*)d_in[br == 0 ? 0 : 1];
    const float* ew = (const float*)d_in[br == 0 ? 2 : 3];
    const int*   ei = (const int*)  d_in[br == 0 ? 4 : 5];
    const int* src = ei;
    const int* dst = ei + NEDGE;
    const float* P[10];
    for (int j = 0; j < 10; ++j) P[j] = (const float*)d_in[6 + br * 10 + j];
    const float *g1w=P[0], *g1b=P[1], *g2w=P[2], *g2b=P[3];
    const float *l1w=P[4], *l1b=P[5], *l2w=P[6], *l2b=P[7], *l3w=P[8], *l3b=P[9];

    // weights -> bf16
    k_cvt_bf16<<<ceil_div(65536, threads), threads, 0, stream>>>(g1w, g1, 65536);
    k_cvt_bf16<<<ceil_div(65536, threads), threads, 0, stream>>>(g2w, g2, 65536);
    k_cvt_bf16<<<ceil_div(65536, threads), threads, 0, stream>>>(l1w, l1, 65536);
    k_cvt_bf16<<<ceil_div(32768, threads), threads, 0, stream>>>(l2w, l2, 32768);
    k_cvt_bf16<<<ceil_div(8192,  threads), threads, 0, stream>>>(l3w, l3, 8192);
    // x -> bf16
    k_cvt_bf16<<<ceil_div(nC, threads), threads, 0, stream>>>(x, act, nC);

    // degrees -> dinv (shared by both conv layers)
    k_deg_init   <<<ceil_div(NODES, threads), threads, 0, stream>>>(deg, NODES);
    k_deg_scatter<<<ceil_div(NEDGE, threads), threads, 0, stream>>>(dst, ew, deg, NEDGE);
    k_deg_rsqrt  <<<ceil_div(NODES, threads), threads, 0, stream>>>(deg, NODES);

    long long etot = (long long)NEDGE * (INF / 4);
    int eblk = ceil_div(etot, threads);

    // GCN layer 1
    gemm_f32(act, g1, xw, NODES, 256, 256, 256);
    k_agg_self <<<ceil_div(nC, threads), threads, 0, stream>>>(xw, deg, agg, NODES, 256);
    k_agg_edges<<<eblk, threads, 0, stream>>>(src, dst, ew, deg, xw, agg, NEDGE, 256);
    k_bias_relu_bf16<<<ceil_div(nC, threads), threads, 0, stream>>>(agg, g1b, act, NODES, 256);

    // GCN layer 2
    gemm_f32(act, g2, xw, NODES, 256, 256, 256);
    k_agg_self <<<ceil_div(nC, threads), threads, 0, stream>>>(xw, deg, agg, NODES, 256);
    k_agg_edges<<<eblk, threads, 0, stream>>>(src, dst, ew, deg, xw, agg, NEDGE, 256);
    k_bias_relu_bf16<<<ceil_div(nC, threads), threads, 0, stream>>>(agg, g2b, act, NODES, 256);

    // MLP: 256 -> 256 -> 128 -> 64, fused bias+relu, bf16 out
    gemm_mlp(act,  l1, l1b, act2, NODES, 256, 256, 256);
    gemm_mlp(act2, l2, l2b, act,  NODES, 128, 128, 256);
    __bf16* embDst = (br == 0) ? emb_dis : act2;
    gemm_mlp(act,  l3, l3b, embDst, NODES, 64, 64, 128);

    if (br == 1) {
      k_transpose_pad_bf16<<<ceil_div(64 * NPAD, threads), threads, 0, stream>>>(
          act2, drugT, NODES, NPAD, 64);
    }
  }

  // final: out[12000,12000] f32 = emb_dis @ drug^T  (B = drugT row-major [64][12032])
  gemm_f32(emb_dis, drugT, (float*)d_out, NODES, NODES, NPAD, 64);
}
